// MoE_8074538516568
// MI455X (gfx1250) — compile-verified, matrix operand order checked by reading, not run on hardware
//
#include <hip/hip_runtime.h>

// ---------------- problem constants ----------------
#define TKN   8192          // B*S tokens
#define DM    2048          // d_model
#define IM    1408          // moe inter dim
#define NE    8             // routed experts
#define TOPK  2
#define SHBASE (2*TKN)      // shared segment base row (routed rows total exactly 2*T)
#define RTOT  (3*TKN)       // total grouped rows (2T routed + T shared)

typedef __attribute__((ext_vector_type(16))) __bf16 v16bf;
typedef __attribute__((ext_vector_type(8)))  float  v8f;
typedef __attribute__((ext_vector_type(4)))  unsigned int v4u;
typedef __attribute__((ext_vector_type(8)))  int v8i;
typedef __attribute__((ext_vector_type(4)))  int v4i;

union FragU { unsigned u[8]; v16bf v; };

__device__ __forceinline__ unsigned short f2bf(float f) {
  unsigned u = __float_as_uint(f);
  u += 0x7FFFu + ((u >> 16) & 1u);          // round-to-nearest-even
  return (unsigned short)(u >> 16);
}

// ---- Tensor Data Mover: 2D tile load (128 rows x 64B) into LDS ----
// D# per CDNA5 ISA ch.8: group0 = {count/type/lds/global}, group1 = dims/strides.
// data_size = 8B units. 6-arg builtin form (extra int32x8 group, then cpol).
__device__ __forceinline__ void tdm_load_tile(const unsigned short* lds_dst,
                                              const unsigned short* gsrc,
                                              unsigned stride_units) {
  unsigned long long ga = (unsigned long long)(uintptr_t)gsrc;
  unsigned lds_off = (unsigned)(uintptr_t)lds_dst;  // low 32 bits = LDS byte addr
  const unsigned td0 = 1u << 20, td1 = 1u << 20;    // generous tensor dims (no OOB)
  v4u g0;
  g0[0] = 1u;                                        // count=1, user descriptor
  g0[1] = lds_off;                                   // lds_addr
  g0[2] = (unsigned)(ga & 0xFFFFFFFFu);              // global_addr[31:0]
  g0[3] = (unsigned)((ga >> 32) & 0x1FFFFFFu) | (2u << 30);  // addr[56:32], type=2
  v8i g1;
  g1[0] = (int)(3u << 16);                           // data_size = 8B
  g1[1] = (int)((td0 & 0xFFFFu) << 16);              // tensor_dim0[15:0]
  g1[2] = (int)(((td0 >> 16) & 0xFFFFu) | ((td1 & 0xFFFFu) << 16));
  g1[3] = (int)(((td1 >> 16) & 0xFFFFu) | (8u << 16));   // tile_dim0 = 8 (64B row)
  g1[4] = (int)128;                                  // tile_dim1 = 128 rows
  g1[5] = (int)stride_units;                         // tensor_dim0_stride[31:0]
  g1[6] = 0;
  g1[7] = 0;
  v4i g2 = {0, 0, 0, 0};
  v4i g3 = {0, 0, 0, 0};
  v8i g4 = {0, 0, 0, 0, 0, 0, 0, 0};
  __builtin_amdgcn_tensor_load_to_lds(g0, g1, g2, g3, g4, 0);
}

// Load one 16x32 bf16 fragment from LDS (tile row-major, stride 32 shorts).
// CDNA5 16-bit operand layout: lane m=L&15, half=L>>4;
// VGPR v holds K = ((v&4)?16:0) + half*8 + (v&3)*2 , +1 (contiguous pair).
__device__ __forceinline__ v16bf load_frag(const unsigned short* tile, int lane) {
  const int m    = lane & 15;
  const int half = lane >> 4;
  const unsigned* p = (const unsigned*)tile;   // 16 dwords per 32-short row
  FragU f;
#pragma unroll
  for (int v = 0; v < 8; ++v) {
    const int kk = ((v & 4) << 2) + half * 8 + ((v & 3) << 1);
    f.u[v] = p[m * 16 + (kk >> 1)];
  }
  return f.v;
}

// ---------------- fp32 -> bf16 convert (row-major copy) ----------------
__global__ void cvt_bf16_kernel(const float* __restrict__ in,
                                unsigned short* __restrict__ out, int n) {
  int i = blockIdx.x * 256 + threadIdx.x;
  if (i < n) out[i] = f2bf(in[i]);
}

// ---- fp32 [K][N] -> bf16 [N][K] transposed convert (per group z) ----
__global__ __launch_bounds__(256) void cvt_transpose_kernel(
    const float* __restrict__ src, unsigned short* __restrict__ dst,
    int K, int N) {
  __shared__ unsigned short tile[32][33];
  const int g = blockIdx.z;
  const float* s = src + (size_t)g * K * N;
  unsigned short* d = dst + (size_t)g * K * N;
  const int n0 = blockIdx.x * 32;
  const int k0 = blockIdx.y * 32;
  const int tx = threadIdx.x & 31;
  const int ty = threadIdx.x >> 5;   // 0..7
#pragma unroll
  for (int r = 0; r < 32; r += 8)
    tile[ty + r][tx] = f2bf(s[(size_t)(k0 + ty + r) * N + n0 + tx]);
  __syncthreads();
#pragma unroll
  for (int r = 0; r < 32; r += 8)
    d[(size_t)(n0 + ty + r) * K + k0 + tx] = tile[tx][ty + r];
}

// ---------------- routing: one block per token ----------------
__global__ __launch_bounds__(256) void routing_kernel(
    const float* __restrict__ x, const float* __restrict__ cent,
    const float* __restrict__ ebias,
    float* __restrict__ aff_out, float* __restrict__ gat_out,
    float* __restrict__ top_out,
    int* __restrict__ topk_i, float* __restrict__ gating_s,
    int* __restrict__ counts) {
  const int t = blockIdx.x;
  __shared__ float red[8 * 256];
  float acc[8] = {0, 0, 0, 0, 0, 0, 0, 0};
  const float* xr = x + (size_t)t * DM;
  for (int d = threadIdx.x; d < DM; d += 256) {
    float xv = xr[d];
#pragma unroll
    for (int e = 0; e < 8; ++e) acc[e] += xv * cent[e * DM + d];
  }
#pragma unroll
  for (int e = 0; e < 8; ++e) red[e * 256 + threadIdx.x] = acc[e];
  __syncthreads();
  for (int s = 128; s > 0; s >>= 1) {
    if (threadIdx.x < s) {
#pragma unroll
      for (int e = 0; e < 8; ++e)
        red[e * 256 + threadIdx.x] += red[e * 256 + threadIdx.x + s];
    }
    __syncthreads();
  }
  if (threadIdx.x == 0) {
    float aff[8], biased[8];
#pragma unroll
    for (int e = 0; e < 8; ++e) {
      aff[e] = 1.0f / (1.0f + expf(-red[e * 256]));
      biased[e] = aff[e] + ebias[e];
      aff_out[(size_t)t * 8 + e] = aff[e];
    }
    int e0 = 0; float m0v = biased[0];
#pragma unroll
    for (int e = 1; e < 8; ++e) if (biased[e] > m0v) { m0v = biased[e]; e0 = e; }
    int e1 = (e0 == 0) ? 1 : 0; float m1v = biased[e1];
#pragma unroll
    for (int e = 0; e < 8; ++e)
      if (e != e0 && biased[e] > m1v) { m1v = biased[e]; e1 = e; }
    float s0 = aff[e0], s1 = aff[e1];
    float mx = fmaxf(s0, s1);
    float x0 = expf(s0 - mx), x1 = expf(s1 - mx);
    float inv = 1.0f / (x0 + x1);
    float g0 = x0 * inv, g1 = x1 * inv;
    gat_out[t * 2 + 0] = g0;  gat_out[t * 2 + 1] = g1;
    top_out[t * 2 + 0] = (float)e0;  top_out[t * 2 + 1] = (float)e1;
    topk_i[t * 2 + 0] = e0;  topk_i[t * 2 + 1] = e1;
    gating_s[t * 2 + 0] = g0; gating_s[t * 2 + 1] = g1;
    atomicAdd(&counts[e0], 1);
    atomicAdd(&counts[e1], 1);
  }
}

__global__ void prefix_kernel(int* counts, int* bases) {
  if (threadIdx.x == 0 && blockIdx.x == 0) {
    int b = 0;
    for (int e = 0; e < 8; ++e) { bases[e] = b; b += counts[e]; }
    bases[8] = SHBASE;
    counts[8] = TKN;
  }
}

__global__ void scatter_kernel(const int* __restrict__ topk_i,
                               const float* __restrict__ gating_s,
                               const int* __restrict__ bases, int* fills,
                               int* __restrict__ tokenlist,
                               float* __restrict__ gates,
                               int* __restrict__ outslot) {
  int t = blockIdx.x * 256 + threadIdx.x;
  if (t >= TKN) return;
#pragma unroll
  for (int k = 0; k < TOPK; ++k) {
    int e = topk_i[t * 2 + k];
    int j = atomicAdd(&fills[e], 1);
    int slot = bases[e] + j;
    tokenlist[slot] = t;
    gates[slot] = gating_s[t * 2 + k];
    outslot[t * 2 + k] = slot;
  }
  tokenlist[SHBASE + t] = t;
  gates[SHBASE + t] = 1.0f;
}

// ---------------- grouped GEMM1: h = silu(Xg@w1+b1)*(Xg@w3+b3) ----------------
// Weights pre-transposed: w1t/w3t are bf16 [group][N=IM][K=DM].
// grid: (IM/128, TKN/128, 9); block 256 = 8 waves; wave -> 32x64 tile.
// B tiles DMA'd by TDM (waves 0/1), double-buffered; A gathered manually.
__global__ __launch_bounds__(256) void gemm1_kernel(
    const unsigned short* __restrict__ xb,
    const unsigned short* __restrict__ w1t,
    const unsigned short* __restrict__ w3t,
    const float* __restrict__ rb1, const float* __restrict__ sb1,
    const float* __restrict__ rb3, const float* __restrict__ sb3,
    const int* __restrict__ tokenlist,
    const int* __restrict__ counts, const int* __restrict__ bases,
    unsigned short* __restrict__ h) {
  const int g = blockIdx.z;
  const int count = counts[g];
  const int base  = bases[g];
  const int m0 = blockIdx.y * 128;
  if (m0 >= count) return;
  const int n0 = blockIdx.x * 128;

  const unsigned short* w1 = w1t + (size_t)g * DM * IM + (size_t)n0 * DM;
  const unsigned short* w3 = w3t + (size_t)g * DM * IM + (size_t)n0 * DM;
  const float* b1p = (g < NE) ? (rb1 + (size_t)g * IM) : sb1;
  const float* b3p = (g < NE) ? (rb3 + (size_t)g * IM) : sb3;

  __shared__ unsigned short As[2][128 * 32];   // [m][k]
  __shared__ unsigned short B1s[2][128 * 32];  // [n][k]
  __shared__ unsigned short B3s[2][128 * 32];

  const int tid  = threadIdx.x;
  const int lane = tid & 31;
  const int wid  = tid >> 5;
  const int wm   = wid >> 1;   // 0..3 -> 32-row strip
  const int wn   = wid & 1;    // 0..1 -> 64-col strip

  const int arow  = tid >> 1;
  const int ahalf = tid & 1;
  const int mrow  = m0 + arow < count ? m0 + arow : count - 1;
  const int atok  = tokenlist[base + mrow];
  const unsigned short* xrow = xb + (size_t)atok * DM;

  v8f accU[2][4], accV[2][4];
#pragma unroll
  for (int mi = 0; mi < 2; ++mi)
#pragma unroll
    for (int ni = 0; ni < 4; ++ni) {
      accU[mi][ni] = (v8f){0, 0, 0, 0, 0, 0, 0, 0};
      accV[mi][ni] = (v8f){0, 0, 0, 0, 0, 0, 0, 0};
    }

  // prologue: tile 0
  if (wid == 0) tdm_load_tile(&B1s[0][0], w1, DM / 4);
  if (wid == 1) tdm_load_tile(&B3s[0][0], w3, DM / 4);
  {
    const uint4* src = (const uint4*)(xrow + ahalf * 16);
    uint4* dst = (uint4*)(&As[0][arow * 32 + ahalf * 16]);
    dst[0] = src[0]; dst[1] = src[1];
  }
  __builtin_amdgcn_s_wait_tensorcnt(0);
  __syncthreads();

  const int KT = DM / 32;
  for (int kt = 0; kt < KT; ++kt) {
    const int cur = kt & 1;
    const int nxt = cur ^ 1;
    const bool more = (kt + 1) < KT;
    uint4 a0, a1;
    if (more) {
      const int k1 = (kt + 1) * 32;
      if (wid == 0) tdm_load_tile(&B1s[nxt][0], w1 + k1, DM / 4);
      if (wid == 1) tdm_load_tile(&B3s[nxt][0], w3 + k1, DM / 4);
      const uint4* src = (const uint4*)(xrow + k1 + ahalf * 16);
      a0 = src[0]; a1 = src[1];
    }
    FragU a[2];
#pragma unroll
    for (int mi = 0; mi < 2; ++mi)
      a[mi].v = load_frag(&As[cur][(wm * 32 + mi * 16) * 32], lane);
#pragma unroll
    for (int ni = 0; ni < 4; ++ni) {
      FragU b1f, b3f;
      b1f.v = load_frag(&B1s[cur][(wn * 64 + ni * 16) * 32], lane);
      b3f.v = load_frag(&B3s[cur][(wn * 64 + ni * 16) * 32], lane);
#pragma unroll
      for (int mi = 0; mi < 2; ++mi) {
        accU[mi][ni] = __builtin_amdgcn_wmma_f32_16x16x32_bf16(
            false, a[mi].v, false, b1f.v, (short)0, accU[mi][ni], false, false);
        accV[mi][ni] = __builtin_amdgcn_wmma_f32_16x16x32_bf16(
            false, a[mi].v, false, b3f.v, (short)0, accV[mi][ni], false, false);
      }
    }
    if (more) {
      uint4* dst = (uint4*)(&As[nxt][arow * 32 + ahalf * 16]);
      dst[0] = a0; dst[1] = a1;
    }
    __builtin_amdgcn_s_wait_tensorcnt(0);
    __syncthreads();
  }

  const int nl = lane & 15;
  const int mh = (lane >> 4) * 8;
#pragma unroll
  for (int mi = 0; mi < 2; ++mi)
#pragma unroll
    for (int ni = 0; ni < 4; ++ni) {
      const int n = n0 + wn * 64 + ni * 16 + nl;
      const float b1v = b1p[n];
      const float b3v = b3p[n];
#pragma unroll
      for (int r = 0; r < 8; ++r) {
        const int m = m0 + wm * 32 + mi * 16 + mh + r;
        if (m < count) {
          float u = accU[mi][ni][r] + b1v;
          float v = accV[mi][ni][r] + b3v;
          float s = u / (1.0f + __expf(-u));   // silu
          h[(size_t)(base + m) * IM + n] = f2bf(s * v);
        }
      }
    }
}

// ---------------- grouped GEMM2: out = gate * (h@w2 + b2) ----------------
// w2t: bf16 [group][N=DM][K=IM]. Same tiling; B via TDM double-buffered.
__global__ __launch_bounds__(256) void gemm2_kernel(
    const unsigned short* __restrict__ h,
    const unsigned short* __restrict__ w2t,
    const float* __restrict__ rb2, const float* __restrict__ sb2,
    const float* __restrict__ gates,
    const int* __restrict__ counts, const int* __restrict__ bases,
    float* __restrict__ outb) {
  const int g = blockIdx.z;
  const int count = counts[g];
  const int base  = bases[g];
  const int m0 = blockIdx.y * 128;
  if (m0 >= count) return;
  const int n0 = blockIdx.x * 128;

  const unsigned short* w2 = w2t + (size_t)g * IM * DM + (size_t)n0 * IM;
  const float* b2p = (g < NE) ? (rb2 + (size_t)g * DM) : sb2;

  __shared__ unsigned short As[2][128 * 32];
  __shared__ unsigned short Bs[2][128 * 32];

  const int tid  = threadIdx.x;
  const int lane = tid & 31;
  const int wid  = tid >> 5;
  const int wm   = wid >> 1;
  const int wn   = wid & 1;

  const int arowl = tid >> 1;
  const int ahalf = tid & 1;
  const int mrow  = m0 + arowl < count ? m0 + arowl : count - 1;
  const unsigned short* hrow = h + (size_t)(base + mrow) * IM;

  v8f acc[2][4];
#pragma unroll
  for (int mi = 0; mi < 2; ++mi)
#pragma unroll
    for (int ni = 0; ni < 4; ++ni)
      acc[mi][ni] = (v8f){0, 0, 0, 0, 0, 0, 0, 0};

  if (wid == 0) tdm_load_tile(&Bs[0][0], w2, IM / 4);
  {
    const uint4* src = (const uint4*)(hrow + ahalf * 16);
    uint4* dst = (uint4*)(&As[0][arowl * 32 + ahalf * 16]);
    dst[0] = src[0]; dst[1] = src[1];
  }
  __builtin_amdgcn_s_wait_tensorcnt(0);
  __syncthreads();

  const int KT = IM / 32;
  for (int kt = 0; kt < KT; ++kt) {
    const int cur = kt & 1;
    const int nxt = cur ^ 1;
    const bool more = (kt + 1) < KT;
    uint4 a0, a1;
    if (more) {
      const int k1 = (kt + 1) * 32;
      if (wid == 0) tdm_load_tile(&Bs[nxt][0], w2 + k1, IM / 4);
      const uint4* src = (const uint4*)(hrow + k1 + ahalf * 16);
      a0 = src[0]; a1 = src[1];
    }
    FragU a[2];
#pragma unroll
    for (int mi = 0; mi < 2; ++mi)
      a[mi].v = load_frag(&As[cur][(wm * 32 + mi * 16) * 32], lane);
#pragma unroll
    for (int ni = 0; ni < 4; ++ni) {
      FragU bf;
      bf.v = load_frag(&Bs[cur][(wn * 64 + ni * 16) * 32], lane);
#pragma unroll
      for (int mi = 0; mi < 2; ++mi)
        acc[mi][ni] = __builtin_amdgcn_wmma_f32_16x16x32_bf16(
            false, a[mi].v, false, bf.v, (short)0, acc[mi][ni], false, false);
    }
    if (more) {
      uint4* dst = (uint4*)(&As[nxt][arowl * 32 + ahalf * 16]);
      dst[0] = a0; dst[1] = a1;
    }
    __builtin_amdgcn_s_wait_tensorcnt(0);
    __syncthreads();
  }

  const int nl = lane & 15;
  const int mh = (lane >> 4) * 8;
#pragma unroll
  for (int mi = 0; mi < 2; ++mi)
#pragma unroll
    for (int ni = 0; ni < 4; ++ni) {
      const int n = n0 + wn * 64 + ni * 16 + nl;
      const float b2v = b2p[n];
#pragma unroll
      for (int r = 0; r < 8; ++r) {
        const int m = m0 + wm * 32 + mi * 16 + mh + r;
        if (m < count) {
          float gate = gates[base + m];
          outb[(size_t)(base + m) * DM + n] = (acc[mi][ni][r] + b2v) * gate;
        }
      }
    }
}

// ---------------- final: hidden = x + shared + routed0 + routed1 ----------------
__global__ void final_kernel(const float* __restrict__ x,
                             const float* __restrict__ outb,
                             const int* __restrict__ outslot,
                             float* __restrict__ hidden) {
  const int CD = DM / 4;
  int idx = blockIdx.x * 256 + threadIdx.x;
  if (idx >= TKN * CD) return;
  int t = idx / CD;
  int c = (idx % CD) * 4;
  int s0 = outslot[t * 2 + 0];
  int s1 = outslot[t * 2 + 1];
  float4 xv = *(const float4*)(x + (size_t)t * DM + c);
  float4 sh = *(const float4*)(outb + (size_t)(SHBASE + t) * DM + c);
  float4 r0 = *(const float4*)(outb + (size_t)s0 * DM + c);
  float4 r1 = *(const float4*)(outb + (size_t)s1 * DM + c);
  float4 o;
  o.x = xv.x + sh.x + r0.x + r1.x;
  o.y = xv.y + sh.y + r0.y + r1.y;
  o.z = xv.z + sh.z + r0.z + r1.z;
  o.w = xv.w + sh.w + r0.w + r1.w;
  *(float4*)(hidden + (size_t)t * DM + c) = o;
}

// ---------------- launch ----------------
extern "C" void kernel_launch(void* const* d_in, const int* in_sizes, int n_in,
                              void* d_out, int out_size, void* d_ws, size_t ws_size,
                              hipStream_t stream) {
  (void)in_sizes; (void)n_in; (void)out_size; (void)ws_size;
  const float* x     = (const float*)d_in[0];
  const float* cent  = (const float*)d_in[1];
  const float* ebias = (const float*)d_in[2];
  const float* sw1   = (const float*)d_in[3];
  const float* sb1   = (const float*)d_in[4];
  const float* sw2   = (const float*)d_in[5];
  const float* sb2   = (const float*)d_in[6];
  const float* sw3   = (const float*)d_in[7];
  const float* sb3   = (const float*)d_in[8];
  const float* rw1   = (const float*)d_in[9];
  const float* rb1   = (const float*)d_in[10];
  const float* rw2   = (const float*)d_in[11];
  const float* rb2   = (const float*)d_in[12];
  const float* rw3   = (const float*)d_in[13];
  const float* rb3   = (const float*)d_in[14];

  char* ws = (char*)d_ws;
  size_t off = 0;
  auto bump = [&](size_t bytes) -> void* {
    off = (off + 255) & ~(size_t)255;
    void* p = ws + off;
    off += bytes;
    return p;
  };
  unsigned short* xb  = (unsigned short*)bump((size_t)TKN * DM * 2);
  unsigned short* w1t = (unsigned short*)bump((size_t)9 * DM * IM * 2);
  unsigned short* w3t = (unsigned short*)bump((size_t)9 * DM * IM * 2);
  unsigned short* w2t = (unsigned short*)bump((size_t)9 * IM * DM * 2);
  unsigned short* hb  = (unsigned short*)bump((size_t)RTOT * IM * 2);
  float* outb    = (float*)bump((size_t)RTOT * DM * 4);
  int* tokenlist = (int*)bump((size_t)RTOT * 4);
  float* gates   = (float*)bump((size_t)RTOT * 4);
  int* outslot   = (int*)bump((size_t)TKN * 2 * 4);
  int* topk_i    = (int*)bump((size_t)TKN * 2 * 4);
  float* gating_s= (float*)bump((size_t)TKN * 2 * 4);
  int* counts    = (int*)bump(16 * 4);
  int* fills     = (int*)bump(16 * 4);
  int* bases     = (int*)bump(16 * 4);

  (void)hipMemsetAsync(counts, 0, 16 * 4, stream);
  (void)hipMemsetAsync(fills, 0, 16 * 4, stream);

  // activations: plain bf16 convert (row-major [token][K])
  {
    long long n = (long long)TKN * DM;
    cvt_bf16_kernel<<<(unsigned)((n + 255) / 256), 256, 0, stream>>>(x, xb, (int)n);
  }
  // weights: convert + transpose to [N][K] so GEMM B-tiles are 2D TDM tiles
  cvt_transpose_kernel<<<dim3(IM / 32, DM / 32, NE), 256, 0, stream>>>(rw1, w1t, DM, IM);
  cvt_transpose_kernel<<<dim3(IM / 32, DM / 32, 1), 256, 0, stream>>>(
      sw1, w1t + (size_t)NE * DM * IM, DM, IM);
  cvt_transpose_kernel<<<dim3(IM / 32, DM / 32, NE), 256, 0, stream>>>(rw3, w3t, DM, IM);
  cvt_transpose_kernel<<<dim3(IM / 32, DM / 32, 1), 256, 0, stream>>>(
      sw3, w3t + (size_t)NE * DM * IM, DM, IM);
  cvt_transpose_kernel<<<dim3(DM / 32, IM / 32, NE), 256, 0, stream>>>(rw2, w2t, IM, DM);
  cvt_transpose_kernel<<<dim3(DM / 32, IM / 32, 1), 256, 0, stream>>>(
      sw2, w2t + (size_t)NE * IM * DM, IM, DM);

  float* out     = (float*)d_out;
  float* hidden  = out;
  float* aff_out = out + (size_t)TKN * DM;
  float* gat_out = aff_out + (size_t)TKN * NE;
  float* top_out = gat_out + (size_t)TKN * TOPK;

  routing_kernel<<<TKN, 256, 0, stream>>>(x, cent, ebias, aff_out, gat_out,
                                          top_out, topk_i, gating_s, counts);
  prefix_kernel<<<1, 32, 0, stream>>>(counts, bases);
  scatter_kernel<<<TKN / 256, 256, 0, stream>>>(topk_i, gating_s, bases, fills,
                                                tokenlist, gates, outslot);
  gemm1_kernel<<<dim3(IM / 128, TKN / 128, 9), 256, 0, stream>>>(
      xb, w1t, w3t, rb1, sb1, rb3, sb3, tokenlist, counts, bases, hb);
  gemm2_kernel<<<dim3(DM / 128, TKN / 128, 9), 256, 0, stream>>>(
      hb, w2t, rb2, sb2, gates, counts, bases, outb);
  final_kernel<<<(TKN * (DM / 4)) / 256, 256, 0, stream>>>(x, outb, outslot, hidden);
}
